// PollutionGNN_83614423318975
// MI455X (gfx1250) — compile-verified, matrix operand order checked by reading, not run on hardware
//
#include <hip/hip_runtime.h>
#include <hip/hip_bf16.h>
#include <math.h>

// ---------------------------------------------------------------------------
// Types for CDNA5 WMMA
// ---------------------------------------------------------------------------
typedef __attribute__((ext_vector_type(16))) __bf16 v16bf;
typedef __attribute__((ext_vector_type(8)))  float  v8f;
typedef __attribute__((ext_vector_type(4)))  float  vf4;
typedef __attribute__((ext_vector_type(4)))  __bf16 v4bf;

#define T_STEPS 24
#define N_NODES 20000
#define F_IN    64
#define HID     256
#define E_EDGES 320000

// GEMM tiling
#define BM 128
#define BN 128
#define BK 32
#define LDT (BK + 8)   // LDS stride (bf16): 40 elems = 80B = 20 banks; 16B-aligned rows

// ---------------------------------------------------------------------------
// WMMA fragment loader (ISA 7.12.2, 16-bit A 16x32 layout; B staged K-major
// per-row so it uses the identical per-lane layout with N as the row index).
//   lanes 0-15 (row=lane):    VGPR v<4 : K = 2v,2v+1   VGPR v>=4: K = 16+2(v-4)..
//   lanes 16-31 (row=lane-16): same with K offset +8
// Each fragment = two contiguous 16B runs -> 2x ds_load_b128 per fragment.
// ---------------------------------------------------------------------------
__device__ inline v16bf pg_load_frag(const __bf16* __restrict__ base, int r, int half) {
  v16bf f;
  const __bf16* p = base + r * LDT + half * 8;
#pragma unroll
  for (int v = 0; v < 8; ++v) {
    int kk = (v < 4) ? (2 * v) : (16 + 2 * (v - 4));
    f[2 * v]     = p[kk];
    f[2 * v + 1] = p[kk + 1];
  }
  return f;
}

// packed f32x4 -> bf16x4 (lowers to v_cvt_pk_bf16_f32 pairs)
__device__ inline uint2 pg_pack4(vf4 v) {
  v4bf b = __builtin_convertvector(v, v4bf);
  return __builtin_bit_cast(uint2, b);
}

// ---------------------------------------------------------------------------
// Tiled GEMM: C[M x N] = A[M x K](f32) * Bt[N x K](bf16, N-major)  (+bias[N])
// 256 threads = 8 waves; wave grid 4x2 over the 128x128 block tile;
// each wave: 2x4 tiles of 16x16 -> 8 v_wmma_f32_16x16x32_bf16 per K-chunk.
// Global loads for K-chunk kc+1 are issued before the WMMA work of chunk kc.
// Staging: thread owns half a row -> A: 4x global b128 + 2x ds b128;
//          B: 2x global b128 + 2x ds b128.
// ---------------------------------------------------------------------------
template <bool BIAS>
__global__ __launch_bounds__(256)
void pg_gemm_bf16(const float* __restrict__ A, const __bf16* __restrict__ Bt,
                  float* __restrict__ C, const float* __restrict__ bias,
                  int M, int N, int K) {
  __shared__ __align__(16) __bf16 As[BM * LDT];
  __shared__ __align__(16) __bf16 Bs[BN * LDT];

  const int tid  = threadIdx.x;
  const int lane = tid & 31;
  const int wave = tid >> 5;
  const int half = lane >> 4;
  const int r    = lane & 15;
  const int waveM = (wave >> 1) * 32;  // 0,32,64,96
  const int waveN = (wave & 1) * 64;   // 0,64
  const int blockM = blockIdx.y * BM;
  const int blockN = blockIdx.x * BN;
  const bool fullM = (blockM + BM) <= M;   // uniform per block

  // Staging coordinates: each thread owns a contiguous 16-element half row.
  const int sm = tid >> 1;             // row (A) / col (B) within tile: 0..127
  const int sk = (tid & 1) * 16;       // k offset: 0 or 16
  const int aRow = blockM + sm;
  const bool aOk = fullM || (aRow < M);
  const float*  aPtr = A  + (long long)aRow * K + sk;
  const __bf16* bPtr = Bt + (long long)(blockN + sm) * K + sk;

  const int nk = K / BK;

  v8f acc[2][4];
#pragma unroll
  for (int mi = 0; mi < 2; ++mi)
#pragma unroll
    for (int ni = 0; ni < 4; ++ni)
      acc[mi][ni] = (v8f)(0.0f);

  vf4   ra[4];
  uint4 rb[2];

  // ---- prefetch K-chunk 0 ----
#pragma unroll
  for (int i = 0; i < 4; ++i)
    ra[i] = aOk ? *(const vf4*)(aPtr + 4 * i) : (vf4)(0.0f);
#pragma unroll
  for (int i = 0; i < 2; ++i)
    rb[i] = *(const uint4*)(bPtr + 8 * i);

  for (int kc = 0; kc < nk; ++kc) {
    // ---- store prefetched regs into LDS ----
    {
      uint2 p0 = pg_pack4(ra[0]), p1 = pg_pack4(ra[1]);
      uint2 p2 = pg_pack4(ra[2]), p3 = pg_pack4(ra[3]);
      *(uint4*)&As[sm * LDT + sk]     = make_uint4(p0.x, p0.y, p1.x, p1.y);
      *(uint4*)&As[sm * LDT + sk + 8] = make_uint4(p2.x, p2.y, p3.x, p3.y);
      *(uint4*)&Bs[sm * LDT + sk]     = rb[0];
      *(uint4*)&Bs[sm * LDT + sk + 8] = rb[1];
    }
    __syncthreads();

    // ---- issue global loads for next K-chunk (hidden behind WMMA) ----
    if (kc + 1 < nk) {
      int k0 = (kc + 1) * BK;
#pragma unroll
      for (int i = 0; i < 4; ++i)
        ra[i] = aOk ? *(const vf4*)(aPtr + k0 + 4 * i) : (vf4)(0.0f);
#pragma unroll
      for (int i = 0; i < 2; ++i)
        rb[i] = *(const uint4*)(bPtr + k0 + 8 * i);
    }

    // ---- WMMA compute on current chunk ----
    v16bf fa[2], fb[4];
#pragma unroll
    for (int mi = 0; mi < 2; ++mi)
      fa[mi] = pg_load_frag(&As[(waveM + 16 * mi) * LDT], r, half);
#pragma unroll
    for (int ni = 0; ni < 4; ++ni)
      fb[ni] = pg_load_frag(&Bs[(waveN + 16 * ni) * LDT], r, half);

#pragma unroll
    for (int mi = 0; mi < 2; ++mi)
#pragma unroll
      for (int ni = 0; ni < 4; ++ni)
        acc[mi][ni] = __builtin_amdgcn_wmma_f32_16x16x32_bf16(
            false, fa[mi], false, fb[ni], (short)0, acc[mi][ni], false, false);
    __syncthreads();
  }

  // Store D: VGPR vr -> M = vr + 8*half, N = r   (ISA 7.12.2 C/D layout)
#pragma unroll
  for (int mi = 0; mi < 2; ++mi) {
#pragma unroll
    for (int ni = 0; ni < 4; ++ni) {
      int col = blockN + waveN + ni * 16 + r;
      float bv = BIAS ? bias[col] : 0.0f;
#pragma unroll
      for (int vr = 0; vr < 8; ++vr) {
        int row = blockM + waveM + mi * 16 + vr + 8 * half;
        if (fullM || row < M) C[(long long)row * N + col] = acc[mi][ni][vr] + bv;
      }
    }
  }
}

// ---------------------------------------------------------------------------
// Prep kernels
// ---------------------------------------------------------------------------
__global__ void pg_cvt_bf16(const float* __restrict__ in, __bf16* __restrict__ out, int n) {
  int i = blockIdx.x * 256 + threadIdx.x;
  if (i < n) out[i] = (__bf16)in[i];
}

// in: [R][C] f32 ; out: [C][R] bf16   (K x N weight -> N x K bf16)
__global__ void pg_cvt_transpose_bf16(const float* __restrict__ in, __bf16* __restrict__ out,
                                      int R, int C) {
  int i = blockIdx.x * 256 + threadIdx.x;
  if (i < R * C) {
    int rr = i / C, cc = i - rr * C;
    out[(long long)cc * R + rr] = (__bf16)in[i];
  }
}

__global__ void pg_fill(float* __restrict__ p, float v, int n) {
  int i = blockIdx.x * 256 + threadIdx.x;
  if (i < n) p[i] = v;
}

__global__ void pg_deg_scatter(const int* __restrict__ dst, const float* __restrict__ ew,
                               float* __restrict__ deg, int E_) {
  int e = blockIdx.x * 256 + threadIdx.x;
  if (e < E_) unsafeAtomicAdd(&deg[dst[e]], ew[e]);
}

__global__ void pg_dinv(const float* __restrict__ deg, float* __restrict__ dinv,
                        float* __restrict__ nself, int n) {
  int i = blockIdx.x * 256 + threadIdx.x;
  if (i < n) {
    float d = deg[i];
    float v = (d > 0.0f) ? rsqrtf(d) : 0.0f;
    dinv[i]  = v;
    nself[i] = v * v;   // self-loop weight 1.0
  }
}

__global__ void pg_norm_e(const int* __restrict__ src, const int* __restrict__ dst,
                          const float* __restrict__ ew, const float* __restrict__ dinv,
                          float* __restrict__ norm, int E_) {
  int e = blockIdx.x * 256 + threadIdx.x;
  if (e < E_) norm[e] = dinv[src[e]] * ew[e] * dinv[dst[e]];
}

// ---------------------------------------------------------------------------
// Aggregation: agg = D^-1/2 A_hat D^-1/2 * h
// ---------------------------------------------------------------------------
__global__ void pg_agg_init(const float* __restrict__ h, const float* __restrict__ nself,
                            float* __restrict__ agg, int nelem) {
  int i = blockIdx.x * 256 + threadIdx.x;
  if (i < nelem) agg[i] = nself[i >> 8] * h[i];
}

__global__ void pg_agg_scatter(const float* __restrict__ h, const int* __restrict__ src,
                               const int* __restrict__ dst, const float* __restrict__ norm,
                               float* __restrict__ agg, int E_) {
  long long idx = (long long)blockIdx.x * 256 + threadIdx.x;
  int e = (int)(idx >> 6);
  if (e >= E_) return;
  int f = ((int)idx & 63) * 4;
  int s = src[e], d = dst[e];
  float w = norm[e];
  const float4 hv = *(const float4*)&h[(long long)s * HID + f];
  float* a = &agg[(long long)d * HID + f];
  unsafeAtomicAdd(a + 0, w * hv.x);
  unsafeAtomicAdd(a + 1, w * hv.y);
  unsafeAtomicAdd(a + 2, w * hv.z);
  unsafeAtomicAdd(a + 3, w * hv.w);
}

__global__ void pg_bias_leaky(const float* __restrict__ agg, const float* __restrict__ b,
                              float* __restrict__ out, int nelem) {
  int i = blockIdx.x * 256 + threadIdx.x;
  if (i < nelem) {
    float v = agg[i] + b[i & 255];
    out[i] = (v >= 0.0f) ? v : 0.01f * v;
  }
}

// ---------------------------------------------------------------------------
// LayerNorm over H=256: one wave per node (wave32), 8 nodes per block
// ---------------------------------------------------------------------------
__device__ inline float pg_wred(float v) {
#pragma unroll
  for (int o = 16; o > 0; o >>= 1) v += __shfl_xor(v, o, 32);
  return v;
}

__global__ __launch_bounds__(256)
void pg_layernorm(const float* __restrict__ agg, const float* __restrict__ b2,
                  const float* __restrict__ lw, const float* __restrict__ lb,
                  float* __restrict__ out, int n_nodes) {
  int node = blockIdx.x * 8 + (threadIdx.x >> 5);
  int lane = threadIdx.x & 31;
  if (node >= n_nodes) return;
  const float* row = &agg[(long long)node * HID];
  float vals[8];
  float s = 0.0f;
#pragma unroll
  for (int j = 0; j < 8; ++j) {
    int f = lane + 32 * j;
    vals[j] = row[f] + b2[f];
    s += vals[j];
  }
  float mu = pg_wred(s) * (1.0f / HID);
  float sq = 0.0f;
#pragma unroll
  for (int j = 0; j < 8; ++j) { float d = vals[j] - mu; sq += d * d; }
  float inv = rsqrtf(pg_wred(sq) * (1.0f / HID) + 1e-5f);
  float* orow = &out[(long long)node * HID];
#pragma unroll
  for (int j = 0; j < 8; ++j) {
    int f = lane + 32 * j;
    orow[f] = (vals[j] - mu) * inv * lw[f] + lb[f];
  }
}

// ---------------------------------------------------------------------------
// Fused GRU gate update: h = (1-z)*n + z*h
// ---------------------------------------------------------------------------
__global__ void pg_gru_gate(const float* __restrict__ GI, const float* __restrict__ GH,
                            float* __restrict__ h, int nelem) {
  int i = blockIdx.x * 256 + threadIdx.x;
  if (i >= nelem) return;
  int n = i >> 8, f = i & 255;
  long long b = (long long)n * (3 * HID) + f;
  float ir = GI[b], iz = GI[b + HID], in_ = GI[b + 2 * HID];
  float hr = GH[b], hz = GH[b + HID], hn  = GH[b + 2 * HID];
  float rg = 1.0f / (1.0f + expf(-(ir + hr)));
  float zg = 1.0f / (1.0f + expf(-(iz + hz)));
  float ng = tanhf(in_ + rg * hn);
  h[i] = (1.0f - zg) * ng + zg * h[i];
}

// ---------------------------------------------------------------------------
// out[n] = h[n,:] . Wfc[0,:] + bfc   (wave per node)
// ---------------------------------------------------------------------------
__global__ __launch_bounds__(256)
void pg_fc(const float* __restrict__ h, const float* __restrict__ Wfc,
           const float* __restrict__ bfc, float* __restrict__ out, int n_nodes) {
  int node = blockIdx.x * 8 + (threadIdx.x >> 5);
  int lane = threadIdx.x & 31;
  if (node >= n_nodes) return;
  const float* row = &h[(long long)node * HID];
  float s = 0.0f;
#pragma unroll
  for (int j = 0; j < 8; ++j) { int f = lane + 32 * j; s += row[f] * Wfc[f]; }
  s = pg_wred(s);
  if (lane == 0) out[node] = s + bfc[0];
}

// ---------------------------------------------------------------------------
// Host launch
// ---------------------------------------------------------------------------
static inline int pg_cdiv(long long a, long long b) { return (int)((a + b - 1) / b); }

extern "C" void kernel_launch(void* const* d_in, const int* in_sizes, int n_in,
                              void* d_out, int out_size, void* d_ws, size_t ws_size,
                              hipStream_t stream) {
  const float* x    = (const float*)d_in[0];   // [T,N,64]
  const int*   ei   = (const int*)d_in[1];     // [2,E]
  const float* ew   = (const float*)d_in[2];   // [E]
  const float* W1   = (const float*)d_in[3];   // [64,256]  (K x N)
  const float* b1   = (const float*)d_in[4];
  const float* W2   = (const float*)d_in[5];   // [256,256] (K x N)
  const float* b2   = (const float*)d_in[6];
  const float* lnw  = (const float*)d_in[7];
  const float* lnb  = (const float*)d_in[8];
  const float* Wih  = (const float*)d_in[9];   // [768,256] = N x K already
  const float* Whh  = (const float*)d_in[10];  // [768,256] = N x K already
  const float* bih  = (const float*)d_in[11];
  const float* bhh  = (const float*)d_in[12];
  const float* Wfc  = (const float*)d_in[13];  // [1,256]
  const float* bfc  = (const float*)d_in[14];
  float* out = (float*)d_out;

  const int* src = ei;             // edge_index[0]
  const int* dst = ei + E_EDGES;   // edge_index[1]

  // Workspace carve-out
  size_t off = 0;
  auto carve = [&](size_t bytes) {
    void* p = (char*)d_ws + off;
    off += (bytes + 255) & ~(size_t)255;
    return p;
  };
  __bf16* W1t   = (__bf16*)carve((size_t)F_IN * HID * 2);       // N x K
  __bf16* W2t   = (__bf16*)carve((size_t)HID * HID * 2);        // N x K
  __bf16* WihB  = (__bf16*)carve((size_t)3 * HID * HID * 2);    // N x K
  __bf16* WhhB  = (__bf16*)carve((size_t)3 * HID * HID * 2);    // N x K
  float* deg    = (float*)carve((size_t)N_NODES * 4);
  float* dinv   = (float*)carve((size_t)N_NODES * 4);
  float* nself  = (float*)carve((size_t)N_NODES * 4);
  float* normE  = (float*)carve((size_t)E_EDGES * 4);
  float* B1     = (float*)carve((size_t)N_NODES * HID * 4);
  float* B2     = (float*)carve((size_t)N_NODES * HID * 4);
  float* B3     = (float*)carve((size_t)N_NODES * HID * 4);
  float* hstate = (float*)carve((size_t)N_NODES * HID * 4);
  float* GI     = (float*)carve((size_t)N_NODES * 3 * HID * 4);
  float* GH     = (float*)carve((size_t)N_NODES * 3 * HID * 4);

  const int NH = N_NODES * HID;
  const dim3 blk(256);

  // Weight conversion (once per launch): all B matrices stored N x K bf16
  pg_cvt_transpose_bf16<<<pg_cdiv(F_IN * HID, 256), blk, 0, stream>>>(W1, W1t, F_IN, HID);
  pg_cvt_transpose_bf16<<<pg_cdiv(HID * HID, 256), blk, 0, stream>>>(W2, W2t, HID, HID);
  pg_cvt_bf16<<<pg_cdiv(3 * HID * HID, 256), blk, 0, stream>>>(Wih, WihB, 3 * HID * HID);
  pg_cvt_bf16<<<pg_cdiv(3 * HID * HID, 256), blk, 0, stream>>>(Whh, WhhB, 3 * HID * HID);

  // GCN normalization (deg init 1.0 = self-loop weight)
  pg_fill<<<pg_cdiv(N_NODES, 256), blk, 0, stream>>>(deg, 1.0f, N_NODES);
  pg_deg_scatter<<<pg_cdiv(E_EDGES, 256), blk, 0, stream>>>(dst, ew, deg, E_EDGES);
  pg_dinv<<<pg_cdiv(N_NODES, 256), blk, 0, stream>>>(deg, dinv, nself, N_NODES);
  pg_norm_e<<<pg_cdiv(E_EDGES, 256), blk, 0, stream>>>(src, dst, ew, dinv, normE, E_EDGES);

  // h0 = 0
  pg_fill<<<pg_cdiv(NH, 256), blk, 0, stream>>>(hstate, 0.0f, NH);

  const dim3 gemm_blk(256);
  const dim3 grid_h(HID / BN, pg_cdiv(N_NODES, BM));        // 2 x 157
  const dim3 grid_g(3 * HID / BN, pg_cdiv(N_NODES, BM));    // 6 x 157
  const int scat_blocks = pg_cdiv((long long)E_EDGES * 64, 256);

  for (int t = 0; t < T_STEPS; ++t) {
    const float* xt = x + (size_t)t * N_NODES * F_IN;
    // GCN layer 1
    pg_gemm_bf16<false><<<grid_h, gemm_blk, 0, stream>>>(xt, W1t, B1, nullptr,
                                                         N_NODES, HID, F_IN);
    pg_agg_init<<<pg_cdiv(NH, 256), blk, 0, stream>>>(B1, nself, B2, NH);
    pg_agg_scatter<<<scat_blocks, blk, 0, stream>>>(B1, src, dst, normE, B2, E_EDGES);
    pg_bias_leaky<<<pg_cdiv(NH, 256), blk, 0, stream>>>(B2, b1, B1, NH);
    // GCN layer 2
    pg_gemm_bf16<false><<<grid_h, gemm_blk, 0, stream>>>(B1, W2t, B3, nullptr,
                                                         N_NODES, HID, HID);
    pg_agg_init<<<pg_cdiv(NH, 256), blk, 0, stream>>>(B3, nself, B2, NH);
    pg_agg_scatter<<<scat_blocks, blk, 0, stream>>>(B3, src, dst, normE, B2, E_EDGES);
    pg_layernorm<<<pg_cdiv(N_NODES, 8), blk, 0, stream>>>(B2, b2, lnw, lnb, B1, N_NODES);
    // GRU step
    pg_gemm_bf16<true><<<grid_g, gemm_blk, 0, stream>>>(B1, WihB, GI, bih,
                                                        N_NODES, 3 * HID, HID);
    pg_gemm_bf16<true><<<grid_g, gemm_blk, 0, stream>>>(hstate, WhhB, GH, bhh,
                                                        N_NODES, 3 * HID, HID);
    pg_gru_gate<<<pg_cdiv(NH, 256), blk, 0, stream>>>(GI, GH, hstate, NH);
  }

  // Output head
  pg_fc<<<pg_cdiv(N_NODES, 8), blk, 0, stream>>>(hstate, Wfc, bfc, out, N_NODES);
}
